// TransformerBlock_69922067579129
// MI455X (gfx1250) — matrix operand order repaired and checked by reading, not compiled, not run on hardware
//
#include <hip/hip_runtime.h>
#include <hip/hip_bf16.h>
#include <math.h>

// ---------------------------------------------------------------------------
// MI455X (gfx1250) transformer block, bf16 WMMA (v_wmma_f32_16x16x32_bf16).
// Compute-bound (~138 GFLOP); all bf16 weights (25MB) + activations fit in
// the 192MB L2, so GEMMs load WMMA fragments directly from global memory with
// contiguous 16B b128 loads (weights pre-transposed so B-fragments are
// contiguous). Attention is flash-style with online softmax using DPP16
// row_ror reductions (no LDS pipeline traffic), and P staged through a
// per-wave LDS tile (same-wave DS ordering, no block barrier) for the
// C-layout -> A-layout conversion.
// ---------------------------------------------------------------------------

typedef __bf16 bf16_t;
typedef bf16_t v16bf __attribute__((ext_vector_type(16)));
typedef float  v8f   __attribute__((ext_vector_type(8)));

union Frag {
  v16bf v;
  uint4 q[2];
};

__device__ __forceinline__ unsigned short f2bf(float f) {
  unsigned int u = __float_as_uint(f);
  u += 0x7fffu + ((u >> 16) & 1u);          // round-to-nearest-even
  return (unsigned short)(u >> 16);
}

__device__ __forceinline__ v8f wmma_bf16(const Frag& a, const Frag& b, v8f c) {
  return __builtin_amdgcn_wmma_f32_16x16x32_bf16(false, a.v, false, b.v,
                                                 (short)0, c, false, false);
}

// DPP16 rotate within the 16-lane row: ctrl = 0x120 | N  (row_ror:N)
template <int CTRL>
__device__ __forceinline__ float dpp_rot(float x) {
  int i = __float_as_int(x);
  int r = __builtin_amdgcn_update_dpp(i, i, CTRL, 0xF, 0xF, false);
  return __int_as_float(r);
}
// Full 16-lane reductions via row_ror 8/4/2/1 (rows of a WMMA C tile live
// entirely inside one 16-lane half, so ring rotation reduces the whole row).
__device__ __forceinline__ float rowmax16(float t) {
  t = fmaxf(t, dpp_rot<0x128>(t));
  t = fmaxf(t, dpp_rot<0x124>(t));
  t = fmaxf(t, dpp_rot<0x122>(t));
  t = fmaxf(t, dpp_rot<0x121>(t));
  return t;
}
__device__ __forceinline__ float rowsum16(float t) {
  t += dpp_rot<0x128>(t);
  t += dpp_rot<0x124>(t);
  t += dpp_rot<0x122>(t);
  t += dpp_rot<0x121>(t);
  return t;
}

// ---------------------------------------------------------------------------
// Weight transpose + fp32->bf16 cast:  Wt[n][k] = bf16(W[k][n])
// ---------------------------------------------------------------------------
__global__ __launch_bounds__(256)
void k_transpose_cast(const float* __restrict__ W, unsigned short* __restrict__ Wt,
                      int K, int N) {
  __shared__ float tile[32][33];
  int n0 = blockIdx.x * 32, k0 = blockIdx.y * 32;
  int tx = threadIdx.x, ty = threadIdx.y;      // (32, 8)
#pragma unroll
  for (int i = 0; i < 4; ++i)
    tile[ty + 8 * i][tx] = W[(size_t)(k0 + ty + 8 * i) * N + n0 + tx];
  __syncthreads();
#pragma unroll
  for (int i = 0; i < 4; ++i)
    Wt[(size_t)(n0 + ty + 8 * i) * K + k0 + tx] = f2bf(tile[tx][ty + 8 * i]);
}

// ---------------------------------------------------------------------------
// RMSNorm (reference: scale * x / (sqrt(mean(x^2)) + eps)), fp32 -> bf16
// One 256-thread block per row of D=1024.
// ---------------------------------------------------------------------------
__global__ __launch_bounds__(256)
void k_rmsnorm(const float* __restrict__ x, const float* __restrict__ sc,
               unsigned short* __restrict__ y) {
  const int D = 1024;
  int row = blockIdx.x;
  const float* xr = x + (size_t)row * D;
  int tid = threadIdx.x, lane = tid & 31, wave = tid >> 5;
  float vals[4];
  float ss = 0.f;
#pragma unroll
  for (int i = 0; i < 4; ++i) {
    float v = xr[tid + 256 * i];
    vals[i] = v;
    ss += v * v;
  }
#pragma unroll
  for (int m = 16; m >= 1; m >>= 1) ss += __shfl_xor(ss, m);
  __shared__ float red[8];
  if (lane == 0) red[wave] = ss;
  __syncthreads();
  float tot = 0.f;
#pragma unroll
  for (int w = 0; w < 8; ++w) tot += red[w];
  float inv = 1.0f / (sqrtf(tot * (1.0f / 1024.0f)) + 1e-5f);
#pragma unroll
  for (int i = 0; i < 4; ++i) {
    int c = tid + 256 * i;
    y[(size_t)row * D + c] = f2bf(sc[c] * vals[i] * inv);
  }
}

// ---------------------------------------------------------------------------
// GEMM: C[M,N] = A[M,K](bf16) @ Bt[N,K]^T(bf16) + bias, epilogue flags.
// Block = 256 threads = 8 waves (2M x 4N); wave tile 32x64; k-step 32.
// Fragments loaded straight from global (L2-resident), all b128 contiguous.
// ---------------------------------------------------------------------------
template <bool BIAS, bool GELU, bool RES, bool OUTBF16>
__global__ __launch_bounds__(256)
void k_gemm(const unsigned short* __restrict__ A, const unsigned short* __restrict__ Bt,
            const float* __restrict__ bias, const float* __restrict__ res,
            void* __restrict__ Cout, int M, int N, int K) {
  int lane = threadIdx.x & 31;
  int wave = threadIdx.x >> 5;
  int wm = wave >> 2, wn = wave & 3;
  int m0 = blockIdx.y * 64 + wm * 32;
  int n0 = blockIdx.x * 256 + wn * 64;
  int lhi = lane >> 4, llo = lane & 15;

  v8f c[2][4] = {};

  for (int ks = 0; ks < K; ks += 32) {
    Frag a[2], b[4];
#pragma unroll
    for (int mi = 0; mi < 2; ++mi) {
      const unsigned short* ap = A + (size_t)(m0 + mi * 16 + llo) * K + ks + lhi * 8;
      a[mi].q[0] = *(const uint4*)ap;
      a[mi].q[1] = *(const uint4*)(ap + 16);
      __builtin_prefetch(ap + 32, 0, 3);      // global_prefetch_b8, next k-step
    }
#pragma unroll
    for (int ni = 0; ni < 4; ++ni) {
      const unsigned short* bp = Bt + (size_t)(n0 + ni * 16 + llo) * K + ks + lhi * 8;
      b[ni].q[0] = *(const uint4*)bp;
      b[ni].q[1] = *(const uint4*)(bp + 16);
      __builtin_prefetch(bp + 32, 0, 3);
    }
#pragma unroll
    for (int mi = 0; mi < 2; ++mi)
#pragma unroll
      for (int ni = 0; ni < 4; ++ni)
        c[mi][ni] = wmma_bf16(a[mi], b[ni], c[mi][ni]);
  }

#pragma unroll
  for (int mi = 0; mi < 2; ++mi) {
#pragma unroll
    for (int ni = 0; ni < 4; ++ni) {
      int n = n0 + ni * 16 + llo;
      float bv = BIAS ? bias[n] : 0.f;
#pragma unroll
      for (int v = 0; v < 8; ++v) {
        int m = m0 + mi * 16 + v + 8 * lhi;
        float val = c[mi][ni][v] + bv;
        if (GELU) val = 0.5f * val * (1.0f + erff(val * 0.70710678118654752f));
        if (RES)  val += res[(size_t)m * N + n];
        if (OUTBF16)
          ((unsigned short*)Cout)[(size_t)m * N + n] = f2bf(val);
        else
          ((float*)Cout)[(size_t)m * N + n] = val;
      }
    }
  }
}

// ---------------------------------------------------------------------------
// Split fused QKV (fp32, [B*T, 3*1024]) into bf16:
//   Q [B,H,T,64] (scaled by 1/sqrt(64)), K [B,H,T,64], V^T [B,H,64,T]
// ---------------------------------------------------------------------------
__global__ __launch_bounds__(256)
void k_split(const float* __restrict__ qkv, unsigned short* __restrict__ q,
             unsigned short* __restrict__ k, unsigned short* __restrict__ vt) {
  const int T = 2048;
  int idx = blockIdx.x * 256 + threadIdx.x;      // over B*T*D
  int bt = idx >> 10;
  int j  = idx & 1023;                            // h*64 + d
  int h  = j >> 6, d = j & 63;
  int b  = bt >> 11, t = bt & 2047;
  const float* row = qkv + (size_t)bt * 3072;
  int bh = b * 16 + h;
  size_t o = ((size_t)bh * T + t) * 64 + d;
  q[o] = f2bf(row[j] * 0.125f);                   // fold 1/sqrt(HD)
  k[o] = f2bf(row[1024 + j]);
  vt[((size_t)bh * 64 + d) * T + t] = f2bf(row[2048 + j]);
}

// ---------------------------------------------------------------------------
// Flash attention. Grid (B*H, T/128); 8 waves/block, each wave owns 16 query
// rows, iterates 32-key blocks with online softmax. 8 WMMAs per key block.
// Per-wave LDS P tile; same-wave DS ordering (DScnt in-order) means only a
// wavefront-scope fence is needed — no block barrier in the hot loop.
// Out: bf16 [B*T, D] row-major (feeds Wo GEMM as A).
// ---------------------------------------------------------------------------
__global__ __launch_bounds__(256)
void k_attn(const unsigned short* __restrict__ Q, const unsigned short* __restrict__ Kk,
            const unsigned short* __restrict__ Vt, unsigned short* __restrict__ Out) {
  const int T = 2048, D = 1024;
  int bh = blockIdx.x;
  int lane = threadIdx.x & 31, wave = threadIdx.x >> 5;
  int lhi = lane >> 4, llo = lane & 15;
  int q0 = blockIdx.y * 128 + wave * 16;

  const unsigned short* Qb = Q  + (size_t)bh * T * 64;
  const unsigned short* Kb = Kk + (size_t)bh * T * 64;
  const unsigned short* Vb = Vt + (size_t)bh * 64 * T;

  __shared__ __align__(16) unsigned short psh[8][16][32];

  Frag qf[2];
#pragma unroll
  for (int s = 0; s < 2; ++s) {
    const unsigned short* qp = Qb + (size_t)(q0 + llo) * 64 + s * 32 + lhi * 8;
    qf[s].q[0] = *(const uint4*)qp;
    qf[s].q[1] = *(const uint4*)(qp + 16);
  }

  float mrun[8], lrun[8];
  v8f o[4] = {};
#pragma unroll
  for (int v = 0; v < 8; ++v) { mrun[v] = -1e30f; lrun[v] = 0.f; }

  for (int kb = 0; kb < T; kb += 32) {
    // S = Q @ K^T  (16 x 32 scores, two 16x16 C tiles)
    v8f st[2] = {};
#pragma unroll
    for (int jt = 0; jt < 2; ++jt) {
#pragma unroll
      for (int s = 0; s < 2; ++s) {
        Frag kf;
        const unsigned short* kp =
            Kb + (size_t)(kb + jt * 16 + llo) * 64 + s * 32 + lhi * 8;
        kf.q[0] = *(const uint4*)kp;
        kf.q[1] = *(const uint4*)(kp + 16);
        st[jt] = wmma_bf16(qf[s], kf, st[jt]);
      }
    }

    // Online softmax: DPP16 row_ror reductions within the 16-lane half.
    float pv[2][8], rescale[8];
#pragma unroll
    for (int v = 0; v < 8; ++v) {
      float t = rowmax16(fmaxf(st[0][v], st[1][v]));
      float mn = fmaxf(mrun[v], t);
      float sc = __expf(mrun[v] - mn);
      float p0 = __expf(st[0][v] - mn);
      float p1 = __expf(st[1][v] - mn);
      float rs = rowsum16(p0 + p1);
      lrun[v] = lrun[v] * sc + rs;
      mrun[v] = mn;
      rescale[v] = sc;
      pv[0][v] = p0;
      pv[1][v] = p1;
    }
#pragma unroll
    for (int dt = 0; dt < 4; ++dt)
#pragma unroll
      for (int v = 0; v < 8; ++v) o[dt][v] *= rescale[v];

    // C-layout -> A-layout for P via this wave's private LDS tile. DS ops
    // from one wave are architecturally in-order; the fences only stop the
    // compiler from reordering the cross-lane LDS accesses.
    __builtin_amdgcn_fence(__ATOMIC_RELEASE, "wavefront");
#pragma unroll
    for (int jt = 0; jt < 2; ++jt)
#pragma unroll
      for (int v = 0; v < 8; ++v)
        psh[wave][v + 8 * lhi][jt * 16 + llo] = f2bf(pv[jt][v]);
    __builtin_amdgcn_fence(__ATOMIC_ACQ_REL, "wavefront");
    Frag pf;
    pf.q[0] = *(const uint4*)&psh[wave][llo][lhi * 8];
    pf.q[1] = *(const uint4*)&psh[wave][llo][16 + lhi * 8];
    __builtin_amdgcn_fence(__ATOMIC_ACQUIRE, "wavefront");

    // O += P @ V  (V^T layout makes B-fragments contiguous)
#pragma unroll
    for (int dt = 0; dt < 4; ++dt) {
      Frag vf;
      const unsigned short* vp = Vb + (size_t)(dt * 16 + llo) * T + kb + lhi * 8;
      vf.q[0] = *(const uint4*)vp;
      vf.q[1] = *(const uint4*)(vp + 16);
      o[dt] = wmma_bf16(pf, vf, o[dt]);
    }
  }

  int b = bh >> 4, h = bh & 15;
#pragma unroll
  for (int dt = 0; dt < 4; ++dt)
#pragma unroll
    for (int v = 0; v < 8; ++v) {
      int t = q0 + v + 8 * lhi;
      int d = dt * 16 + llo;
      Out[((size_t)(b * T + t)) * D + h * 64 + d] = f2bf(o[dt][v] / lrun[v]);
    }
}

// ---------------------------------------------------------------------------
// Host orchestration
// ---------------------------------------------------------------------------
extern "C" void kernel_launch(void* const* d_in, const int* in_sizes, int n_in,
                              void* d_out, int out_size, void* d_ws, size_t ws_size,
                              hipStream_t stream) {
  const float* x      = (const float*)d_in[0];
  const float* scale1 = (const float*)d_in[1];
  const float* scale2 = (const float*)d_in[2];
  const float* Wqkv   = (const float*)d_in[3];
  const float* bqkv   = (const float*)d_in[4];
  const float* Wo     = (const float*)d_in[5];
  const float* bo     = (const float*)d_in[6];
  const float* W1     = (const float*)d_in[7];
  const float* b1     = (const float*)d_in[8];
  const float* W2     = (const float*)d_in[9];
  const float* b2     = (const float*)d_in[10];
  float* out = (float*)d_out;

  const int Tt = 2048, Dv = 1024, D3 = 3072, DFF = 4096, BT = 4096; // B*T
  const size_t MB = 1ull << 20;
  char* ws = (char*)d_ws;

  size_t o_x1   = 0;              // 8MB  bf16 [BT,D]
  size_t o_wqkT = o_x1   + 8*MB;  // 6MB  bf16 [3D,D]
  size_t o_woT  = o_wqkT + 6*MB;  // 2MB  bf16 [D,D]
  size_t o_w1T  = o_woT  + 2*MB;  // 8MB  bf16 [DFF,D]
  size_t o_w2T  = o_w1T  + 8*MB;  // 8MB  bf16 [D,DFF]
  size_t o_qkv  = o_w2T  + 8*MB;  // 48MB f32 [BT,3D]  (reused for ff1 bf16)
  size_t o_q    = o_qkv  + 48*MB; // 8MB  bf16 [B,H,T,64]
  size_t o_k    = o_q    + 8*MB;  // 8MB
  size_t o_vt   = o_k    + 8*MB;  // 8MB  bf16 [B,H,64,T]
  size_t o_attn = o_vt   + 8*MB;  // 8MB  bf16 [BT,D]
  size_t o_x2   = o_attn + 8*MB;  // 16MB f32  [BT,D]
  size_t o_h    = o_x2   + 16*MB; // 8MB  bf16 [BT,D]

  unsigned short* x1    = (unsigned short*)(ws + o_x1);
  unsigned short* wqkvT = (unsigned short*)(ws + o_wqkT);
  unsigned short* woT   = (unsigned short*)(ws + o_woT);
  unsigned short* w1T   = (unsigned short*)(ws + o_w1T);
  unsigned short* w2T   = (unsigned short*)(ws + o_w2T);
  float*          qkvf  = (float*)(ws + o_qkv);
  unsigned short* ff1   = (unsigned short*)(ws + o_qkv);   // reuse after split
  unsigned short* qb    = (unsigned short*)(ws + o_q);
  unsigned short* kbuf  = (unsigned short*)(ws + o_k);
  unsigned short* vtb   = (unsigned short*)(ws + o_vt);
  unsigned short* attnb = (unsigned short*)(ws + o_attn);
  float*          x2    = (float*)(ws + o_x2);
  unsigned short* hb    = (unsigned short*)(ws + o_h);

  dim3 tb(32, 8);
  // weights: fp32 -> bf16, transposed so GEMM B-fragments are contiguous
  k_transpose_cast<<<dim3(D3 / 32, Dv / 32), tb, 0, stream>>>(Wqkv, wqkvT, Dv, D3);
  k_transpose_cast<<<dim3(Dv / 32, Dv / 32), tb, 0, stream>>>(Wo, woT, Dv, Dv);
  k_transpose_cast<<<dim3(DFF / 32, Dv / 32), tb, 0, stream>>>(W1, w1T, Dv, DFF);
  k_transpose_cast<<<dim3(Dv / 32, DFF / 32), tb, 0, stream>>>(W2, w2T, DFF, Dv);

  // x1 = bf16(rmsnorm(x, scale1))
  k_rmsnorm<<<BT, 256, 0, stream>>>(x, scale1, x1);

  // qkv = x1 @ Wqkv + bqkv   (f32 out)
  k_gemm<true, false, false, false><<<dim3(D3 / 256, BT / 64), 256, 0, stream>>>(
      x1, wqkvT, bqkv, nullptr, qkvf, BT, D3, Dv);

  // split to Q (scaled), K, V^T in bf16
  k_split<<<(BT * Dv) / 256, 256, 0, stream>>>(qkvf, qb, kbuf, vtb);

  // flash attention -> attn bf16 [BT, D]
  k_attn<<<dim3(32, Tt / 128), 256, 0, stream>>>(qb, kbuf, vtb, attnb);

  // x2 = x + attn @ Wo + bo   (f32 out)
  k_gemm<true, false, true, false><<<dim3(Dv / 256, BT / 64), 256, 0, stream>>>(
      attnb, woT, bo, x, x2, BT, Dv, Dv);

  // h = bf16(rmsnorm(x2, scale2))
  k_rmsnorm<<<BT, 256, 0, stream>>>(x2, scale2, hb);

  // ff1 = bf16(gelu(h @ W1 + b1))
  k_gemm<true, true, false, true><<<dim3(DFF / 256, BT / 64), 256, 0, stream>>>(
      hb, w1T, b1, nullptr, ff1, BT, DFF, Dv);

  // out = x2 + ff1 @ W2 + b2   (f32 out)
  k_gemm<true, false, true, false><<<dim3(Dv / 256, BT / 64), 256, 0, stream>>>(
      ff1, w2T, b2, x2, out, BT, Dv, DFF);
}
